// ImagePoolingAttn_48893907697804
// MI455X (gfx1250) — matrix operand order repaired and checked by reading, not compiled, not run on hardware
//
#include <hip/hip_runtime.h>
#include <hip/hip_bf16.h>
#include <cstdint>

// ---------------------------------------------------------------------------
// ImagePoolingAttn for MI455X (gfx1250, wave32, WMMA bf16 16x16x32).
// ---------------------------------------------------------------------------
#define EC   256   // embed channels
#define CT   512   // text channels
#define NH   8     // heads
#define HC   32    // head channels (== WMMA K chunk!)
#define NPOS 27    // pooled positions (3 scales x 9 windows)
#define NTXT 512   // text tokens
#define BATCH 32

typedef __attribute__((ext_vector_type(16))) __bf16 v16bf;
typedef __attribute__((ext_vector_type(8)))  __bf16 v8bf;
typedef __attribute__((ext_vector_type(8)))  float  v8f;

__device__ __forceinline__ unsigned short f2bf(float f) {
  unsigned int u = __builtin_bit_cast(unsigned int, f);
  u += 0x7fffu + ((u >> 16) & 1u);              // round-to-nearest-even
  return (unsigned short)(u >> 16);
}

__device__ __forceinline__ v8f vzero() {
  v8f z = {0.f, 0.f, 0.f, 0.f, 0.f, 0.f, 0.f, 0.f};
  return z;
}

// A-fragment (16x32 bf16): lane (g=lane>>4, m=lane&15) holds
//   elem i<8  -> A[m][kbase + g*8 + i]
//   elem i>=8 -> A[m][kbase + 16 + g*8 + (i-8)]
// 'row' must point at A[m][kbase]; both halves are 16B-aligned b128 loads.
__device__ __forceinline__ v16bf load_a16(const unsigned short* row, int g) {
  v8bf lo = *(const v8bf*)(row + g * 8);
  v8bf hi = *(const v8bf*)(row + 16 + g * 8);
  return __builtin_shufflevector(lo, hi, 0, 1, 2, 3, 4, 5, 6, 7,
                                 8, 9, 10, 11, 12, 13, 14, 15);
}

// B-fragment (32x16 bf16): lane (g, n=lane&15) holds elem i<8 -> B[g*8+i][n],
// elem i>=8 -> B[16+g*8+(i-8)][n] (same k order as load_a16).
// Storage is K-contiguous per column: 'col' points at B_T[n][kbase].
__device__ __forceinline__ v16bf load_b16(const unsigned short* col, int g) {
  v8bf lo = *(const v8bf*)(col + g * 8);
  v8bf hi = *(const v8bf*)(col + 16 + g * 8);
  return __builtin_shufflevector(lo, hi, 0, 1, 2, 3, 4, 5, 6, 7,
                                 8, 9, 10, 11, 12, 13, 14, 15);
}

__device__ __forceinline__ v8f wmma_bf16(v16bf a, v16bf b, v8f c) {
  return __builtin_amdgcn_wmma_f32_16x16x32_bf16(
      /*neg_a=*/false, a, /*neg_b=*/false, b,
      /*c_mod=*/(short)0, c, /*reuse_a=*/false, /*reuse_b=*/false);
}

__device__ __forceinline__ float red_max16(float v) {
#pragma unroll
  for (int m = 1; m < 16; m <<= 1) v = fmaxf(v, __shfl_xor(v, m, 32));
  return v;
}
__device__ __forceinline__ float red_sum16(float v) {
#pragma unroll
  for (int m = 1; m < 16; m <<= 1) v += __shfl_xor(v, m, 32);
  return v;
}
__device__ __forceinline__ float red_sum32(float v) {
#pragma unroll
  for (int m = 1; m < 32; m <<= 1) v += __shfl_xor(v, m, 32);
  return v;
}

// ---------------------------------------------------------------------------
// K0a: f32 -> bf16 convert (weights, kept in A layout [M][K]).
// ---------------------------------------------------------------------------
__global__ __launch_bounds__(256) void cvt_kernel(const float* __restrict__ src,
                                                  unsigned short* __restrict__ dst,
                                                  int n) {
  int i = blockIdx.x * 256 + threadIdx.x;
  if (i < n) dst[i] = f2bf(src[i]);
}

// K0b: f32 [R][C] -> bf16 transposed [C][R] (B matrices, K-contiguous cols).
__global__ __launch_bounds__(256) void cvtT_kernel(const float* __restrict__ src,
                                                   unsigned short* __restrict__ dst,
                                                   int n, int cshift, int R) {
  int i = blockIdx.x * 256 + threadIdx.x;
  if (i < n) {
    int r = i >> cshift;
    int c = i & ((1 << cshift) - 1);
    dst[c * R + r] = f2bf(src[i]);
  }
}

// ---------------------------------------------------------------------------
// K1: fused 1x1 conv + adaptive max pool.  One block = (window, batch).
// GEMM  out[256, 16pix] = W[256, Cin] @ x[Cin, 16pix]  over pixel tiles,
// running max kept in WMMA C layout, cross-lane max once at the end.
//
// Software-pipelined: the next tile's global loads (non-temporal, streamed
// once) are issued into VGPRs BEFORE the WMMA loop and only consumed (cvt ->
// LDS back buffer) after it, so HBM traffic overlaps matrix math.  LDS:
// full bf16 weight matrix (<=256KB) + 2 activation tile buffers (<320KB WGP).
// ---------------------------------------------------------------------------
template <int CIN, int HWS, int WIN>
__global__ __launch_bounds__(256) void convpool_kernel(
    const float* __restrict__ x, const unsigned short* __restrict__ wbf,
    const float* __restrict__ bias, float* __restrict__ pooled, int posOff) {
  constexpr int STR = CIN + 8;              // +8 halfwords: bank-spread
  constexpr int HW2 = HWS * HWS;
  constexpr int NPIX = WIN * WIN;
  constexpr int NTILE = NPIX / 16;
  constexpr int NLD = (CIN * 16) / 256;     // f32 loads per thread per tile

  extern __shared__ unsigned short smem[];
  unsigned short* wlds = smem;              // [256][STR] bf16 weights
  unsigned short* abuf0 = smem + 256 * STR; // [16][STR]  tile ping
  unsigned short* abuf1 = abuf0 + 16 * STR; // [16][STR]  tile pong

  const int tid = threadIdx.x;
  const int lane = tid & 31, wid = tid >> 5;
  const int g = lane >> 4, ln16 = lane & 15;
  const int widx = blockIdx.x, b = blockIdx.y;
  const int wy = widx / 3, wx = widx % 3;

  // Stage weights global->LDS, 16B vector copies.
  for (int e = tid * 8; e < 256 * CIN; e += 256 * 8) {
    int m = e / CIN, k = e % CIN;
    *(uint4*)(wlds + m * STR + k) = *(const uint4*)(wbf + e);
  }

  const float* xb = x + (size_t)b * CIN * HW2;

  // element e of a tile -> global offset (pixel n = e&15, channel c = e>>4)
  auto gaddr = [&](int e, int pbase) -> size_t {
    int c = e >> 4, n = e & 15;
    int p = pbase + n;
    int py = p / WIN, px = p % WIN;
    return (size_t)c * HW2 + (wy * WIN + py) * HWS + (wx * WIN + px);
  };

  // Prologue: stage tile 0 directly.
#pragma unroll
  for (int i = 0; i < NLD; ++i) {
    int e = tid + (i << 8);
    int c = e >> 4, n = e & 15;
    abuf0[n * STR + c] = f2bf(__builtin_nontemporal_load(xb + gaddr(e, 0)));
  }
  __syncthreads();

  float rm0[8], rm1[8];
#pragma unroll
  for (int r = 0; r < 8; ++r) { rm0[r] = -3.0e38f; rm1[r] = -3.0e38f; }

  const unsigned short* ar0 = wlds + (wid * 32 + ln16) * STR;   // M-tile 2*wid
  const unsigned short* ar1 = ar0 + 16 * STR;                   // M-tile 2*wid+1

  for (int t = 0; t < NTILE; ++t) {
    unsigned short* cur = (t & 1) ? abuf1 : abuf0;
    unsigned short* nxt = (t & 1) ? abuf0 : abuf1;
    const bool more = (t + 1 < NTILE);

    // Phase A: fire next tile's HBM loads; results land during the GEMM.
    float vals[NLD];
    if (more) {
      int pb = (t + 1) << 4;
#pragma unroll
      for (int i = 0; i < NLD; ++i)
        vals[i] = __builtin_nontemporal_load(xb + gaddr(tid + (i << 8), pb));
    }

    // Phase B: WMMA GEMM on the current tile.
    const unsigned short* brow = cur + ln16 * STR;
    v8f acc0 = vzero(), acc1 = vzero();
    for (int kc = 0; kc < CIN; kc += 32) {
      v16bf bf = load_b16(brow + kc, g);
      v16bf a0 = load_a16(ar0 + kc, g);
      v16bf a1 = load_a16(ar1 + kc, g);
      acc0 = wmma_bf16(a0, bf, acc0);
      acc1 = wmma_bf16(a1, bf, acc1);
    }
#pragma unroll
    for (int r = 0; r < 8; ++r) {
      rm0[r] = fmaxf(rm0[r], acc0[r]);
      rm1[r] = fmaxf(rm1[r], acc1[r]);
    }

    // Phase C: convert + park next tile in the back buffer.
    if (more) {
#pragma unroll
      for (int i = 0; i < NLD; ++i) {
        int e = tid + (i << 8);
        int c = e >> 4, n = e & 15;
        nxt[n * STR + c] = f2bf(vals[i]);
      }
    }
    __syncthreads();
  }

  // Cross-lane max over the 16 pixel columns (per 16-lane half = distinct rows).
  float* prow = pooled + ((size_t)b * NPOS + posOff + widx) * EC;
#pragma unroll
  for (int r = 0; r < 8; ++r) {
    float v0 = red_max16(rm0[r]);
    float v1 = red_max16(rm1[r]);
    if (ln16 == 0) {
      int ch0 = (wid * 2 + 0) * 16 + r + 8 * g;
      int ch1 = (wid * 2 + 1) * 16 + r + 8 * g;
      prow[ch0] = v0 + bias[ch0];   // bias commutes with max
      prow[ch1] = v1 + bias[ch1];
    }
  }
}

// ---------------------------------------------------------------------------
// K2: pooled x -> LayerNorm -> k,v linears.  Tiny (32*27*256), pure VALU.
// ---------------------------------------------------------------------------
__global__ __launch_bounds__(256) void kv_kernel(
    const float* __restrict__ pooled,
    const float* __restrict__ kg, const float* __restrict__ kb2,
    const float* __restrict__ kw, const float* __restrict__ kbias,
    const float* __restrict__ vg, const float* __restrict__ vb2,
    const float* __restrict__ vw, const float* __restrict__ vbias,
    float* __restrict__ kbuf, float* __restrict__ vbuf) {
  __shared__ float xn[NPOS][EC];   // normalized (pre-affine; k/v share it)
  const int tid = threadIdx.x, lane = tid & 31, wid = tid >> 5;
  const int b = blockIdx.x;
  const float* pbase = pooled + (size_t)b * NPOS * EC;

  for (int pos = wid; pos < NPOS; pos += 8) {
    const float* row = pbase + pos * EC;
    float s = 0.f, s2 = 0.f;
    for (int c = lane; c < EC; c += 32) { float v = row[c]; s += v; s2 += v * v; }
    s = red_sum32(s); s2 = red_sum32(s2);
    float mean = s * (1.0f / EC);
    float inv = rsqrtf(s2 * (1.0f / EC) - mean * mean + 1e-5f);
    for (int c = lane; c < EC; c += 32) xn[pos][c] = (row[c] - mean) * inv;
  }
  __syncthreads();

  const int o = tid;  // one output channel per thread
  for (int pos = 0; pos < NPOS; ++pos) {
    float ka = 0.f, va = 0.f;
    for (int c = 0; c < EC; ++c) {
      float nv = xn[pos][c];
      ka += (nv * kg[c] + kb2[c]) * kw[c * EC + o];
      va += (nv * vg[c] + vb2[c]) * vw[c * EC + o];
    }
    kbuf[((size_t)b * NPOS + pos) * EC + o] = ka + kbias[o];
    vbuf[((size_t)b * NPOS + pos) * EC + o] = va + vbias[o];
  }
}

// ---------------------------------------------------------------------------
// K3: q = LN(text) @ q_w + q_b, output bf16 [B][512][256] (A-layout for K4).
// Block = (64 rows, batch); LN rows -> LDS bf16 -> WMMA GEMM vs qwT.
// ---------------------------------------------------------------------------
__global__ __launch_bounds__(256) void q_kernel(
    const float* __restrict__ text,
    const float* __restrict__ lng, const float* __restrict__ lnb,
    const unsigned short* __restrict__ qwT,   // [EC][CT] bf16 (K-contig cols)
    const float* __restrict__ qbias,
    unsigned short* __restrict__ qbf) {
  extern __shared__ unsigned short tln[];     // [64][CT+8] bf16
  const int TSTR = CT + 8;
  const int tid = threadIdx.x, lane = tid & 31, wid = tid >> 5;
  const int g = lane >> 4, ln16 = lane & 15;
  const int mblk = blockIdx.x, b = blockIdx.y;
  const int rbase = mblk * 64;

  for (int r = wid; r < 64; r += 8) {
    const float* row = text + ((size_t)b * NTXT + rbase + r) * CT;
    float s = 0.f, s2 = 0.f;
    for (int c = lane; c < CT; c += 32) { float v = row[c]; s += v; s2 += v * v; }
    s = red_sum32(s); s2 = red_sum32(s2);
    float mean = s * (1.0f / CT);
    float inv = rsqrtf(s2 * (1.0f / CT) - mean * mean + 1e-5f);
    for (int c = lane; c < CT; c += 32)
      tln[r * TSTR + c] = f2bf((row[c] - mean) * inv * lng[c] + lnb[c]);
  }
  __syncthreads();

  v8f acc[4][2];
#pragma unroll
  for (int mi = 0; mi < 4; ++mi)
#pragma unroll
    for (int j = 0; j < 2; ++j) acc[mi][j] = vzero();

  const unsigned short* b0p = qwT + ((wid * 2 + 0) * 16 + ln16) * CT;
  const unsigned short* b1p = qwT + ((wid * 2 + 1) * 16 + ln16) * CT;
  for (int kc = 0; kc < CT; kc += 32) {
    v16bf bf0 = load_b16(b0p + kc, g);
    v16bf bf1 = load_b16(b1p + kc, g);
#pragma unroll
    for (int mi = 0; mi < 4; ++mi) {
      v16bf a = load_a16(&tln[(mi * 16 + ln16) * TSTR + kc], g);
      acc[mi][0] = wmma_bf16(a, bf0, acc[mi][0]);
      acc[mi][1] = wmma_bf16(a, bf1, acc[mi][1]);
    }
  }
#pragma unroll
  for (int mi = 0; mi < 4; ++mi)
#pragma unroll
    for (int j = 0; j < 2; ++j) {
      int o = (wid * 2 + j) * 16 + ln16;
      float bq = qbias[o];
#pragma unroll
      for (int r = 0; r < 8; ++r) {
        int row = rbase + mi * 16 + r + 8 * g;
        qbf[((size_t)b * NTXT + row) * EC + o] = f2bf(acc[mi][j][r] + bq);
      }
    }
}

// ---------------------------------------------------------------------------
// K4: attention.  Block = batch, wave = head (K = HC = 32 = one WMMA chunk).
// 27 keys -> two N-tiles, pad positions masked in softmax; probabilities are
// round-tripped through per-wave LDS (DS is in-order per wave -> no barrier)
// to convert the C-layout into an A-fragment for the P@V WMMA.
// ---------------------------------------------------------------------------
__global__ __launch_bounds__(256) void attn_kernel(
    const unsigned short* __restrict__ qbf,
    const float* __restrict__ kbuf, const float* __restrict__ vbuf,
    unsigned short* __restrict__ attbf) {
  __shared__ unsigned short klds[NH * 32 * 32];  // [head][pos][c]  (B of q@kT)
  __shared__ unsigned short vlds[NH * 32 * 32];  // [head][c][pos]  (B of p@v)
  __shared__ unsigned short plds[NH * 16 * 32];  // [head][row][pos] probs
  const int tid = threadIdx.x, lane = tid & 31, h = tid >> 5;
  const int g = lane >> 4, ln16 = lane & 15;
  const int b = blockIdx.x;

  unsigned short* kh = klds + h * 1024;
  unsigned short* vh = vlds + h * 1024;
  unsigned short* ph = plds + h * 512;

  for (int idx = lane; idx < 1024; idx += 32) {
    int pos = idx >> 5, c = idx & 31;
    float kv = 0.f, vv = 0.f;
    if (pos < NPOS) {
      size_t base = ((size_t)b * NPOS + pos) * EC + h * HC + c;
      kv = kbuf[base]; vv = vbuf[base];
    }
    kh[pos * 32 + c] = f2bf(kv);
    vh[c * 32 + pos] = f2bf(vv);
  }
  __syncthreads();

  v16bf bk0 = load_b16(kh + ln16 * 32, g);          // keys 0..15
  v16bf bk1 = load_b16(kh + (16 + ln16) * 32, g);   // keys 16..31 (pad>=27)
  v16bf bv0 = load_b16(vh + ln16 * 32, g);          // v cols 0..15
  v16bf bv1 = load_b16(vh + (16 + ln16) * 32, g);   // v cols 16..31

  const float scale = 0.17677669529663687f;         // 1/sqrt(HC)
  const bool msk = (ln16 >= NPOS - 16);             // pos = 16+ln16 >= 27

  for (int mt = 0; mt < NTXT / 16; ++mt) {
    const unsigned short* arow =
        qbf + ((size_t)b * NTXT + mt * 16 + ln16) * EC + h * HC;
    v16bf a = load_a16(arow, g);
    v8f s0 = wmma_bf16(a, bk0, vzero());
    v8f s1 = wmma_bf16(a, bk1, vzero());
#pragma unroll
    for (int r = 0; r < 8; ++r) {
      float x0 = s0[r] * scale;
      float x1 = s1[r] * scale;
      float mx = red_max16(fmaxf(x0, msk ? -3.0e38f : x1));
      float e0 = __expf(x0 - mx);
      float e1 = msk ? 0.f : __expf(x1 - mx);
      float sm = red_sum16(e0 + e1);
      float rcp = 1.0f / sm;
      int row = r + 8 * g;
      ph[row * 32 + ln16]      = f2bf(e0 * rcp);
      ph[row * 32 + 16 + ln16] = f2bf(e1 * rcp);
    }
    // Wave-local LDS round-trip: DS ops in-order within a wave.
    v16bf pa = load_a16(ph + ln16 * 32, g);
    v8f o0 = wmma_bf16(pa, bv0, vzero());
    v8f o1 = wmma_bf16(pa, bv1, vzero());
#pragma unroll
    for (int r = 0; r < 8; ++r) {
      int row = mt * 16 + r + 8 * g;
      size_t base = ((size_t)b * NTXT + row) * EC + h * HC;
      attbf[base + ln16]      = f2bf(o0[r]);
      attbf[base + 16 + ln16] = f2bf(o1[r]);
    }
  }
}

// ---------------------------------------------------------------------------
// K5: out = att @ proj_w + proj_b + text.  Block = (32 rows, batch).
// ---------------------------------------------------------------------------
__global__ __launch_bounds__(256) void proj_kernel(
    const unsigned short* __restrict__ attbf,
    const unsigned short* __restrict__ projT,  // [CT][EC] bf16 (K-contig cols)
    const float* __restrict__ pbias,
    const float* __restrict__ text,
    float* __restrict__ out) {
  const int tid = threadIdx.x, lane = tid & 31, wid = tid >> 5;
  const int g = lane >> 4, ln16 = lane & 15;
  const int mblk = blockIdx.x, b = blockIdx.y;
  const int rbase = mblk * 32;

  v8f acc[2][4];
#pragma unroll
  for (int mi = 0; mi < 2; ++mi)
#pragma unroll
    for (int j = 0; j < 4; ++j) acc[mi][j] = vzero();

  const unsigned short* a0 = attbf + ((size_t)b * NTXT + rbase + ln16) * EC;
  const unsigned short* a1 = a0 + 16 * EC;
  for (int kc = 0; kc < EC; kc += 32) {
    v16bf af0 = load_a16(a0 + kc, g);
    v16bf af1 = load_a16(a1 + kc, g);
#pragma unroll
    for (int j = 0; j < 4; ++j) {
      int nt = wid * 4 + j;
      v16bf bf = load_b16(projT + (nt * 16 + ln16) * EC + kc, g);
      acc[0][j] = wmma_bf16(af0, bf, acc[0][j]);
      acc[1][j] = wmma_bf16(af1, bf, acc[1][j]);
    }
  }
#pragma unroll
  for (int mi = 0; mi < 2; ++mi)
#pragma unroll
    for (int j = 0; j < 4; ++j) {
      int t = (wid * 4 + j) * 16 + ln16;
      float pbv = pbias[t];
#pragma unroll
      for (int r = 0; r < 8; ++r) {
        int row = rbase + mi * 16 + r + 8 * g;
        size_t idx = ((size_t)b * NTXT + row) * CT + t;
        out[idx] = acc[mi][j][r] + pbv + text[idx];   // residual
      }
    }
}

// ---------------------------------------------------------------------------
extern "C" void kernel_launch(void* const* d_in, const int* in_sizes, int n_in,
                              void* d_out, int out_size, void* d_ws, size_t ws_size,
                              hipStream_t stream) {
  (void)in_sizes; (void)n_in; (void)out_size; (void)ws_size;
  const float* x0  = (const float*)d_in[0];
  const float* x1  = (const float*)d_in[1];
  const float* x2  = (const float*)d_in[2];
  const float* txt = (const float*)d_in[3];
  const float* cw0 = (const float*)d_in[4];  const float* cb0 = (const float*)d_in[5];
  const float* cw1 = (const float*)d_in[6];  const float* cb1 = (const float*)d_in[7];
  const float* cw2 = (const float*)d_in[8];  const float* cb2 = (const float*)d_in[9];
  const float* qlg = (const float*)d_in[10]; const float* qlb = (const float*)d_in[11];
  const float* qw  = (const float*)d_in[12]; const float* qb  = (const float*)d_in[13];
  const float* klg = (const float*)d_in[14]; const float* klb = (const float*)d_in[15];
  const float* kw  = (const float*)d_in[16]; const float* kb  = (const float*)d_in[17];
  const float* vlg = (const float*)d_in[18]; const float* vlb = (const float*)d_in[19];
  const float* vw  = (const float*)d_in[20]; const float* vb  = (const float*)d_in[21];
  const float* pw  = (const float*)d_in[22]; const float* pb  = (const float*)d_in[23];
  float* out = (float*)d_out;

  char* ws = (char*)d_ws;
  size_t off = 0;
  auto take = [&](size_t bytes) -> char* {
    char* p = ws + off;
    off = (off + bytes + 255) & ~(size_t)255;
    return p;
  };
  unsigned short* wbf0 = (unsigned short*)take((size_t)EC * 256 * 2);
  unsigned short* wbf1 = (unsigned short*)take((size_t)EC * 512 * 2);
  unsigned short* wbf2 = (unsigned short*)take((size_t)EC * 512 * 2);
  unsigned short* qwT  = (unsigned short*)take((size_t)EC * CT * 2);
  unsigned short* prT  = (unsigned short*)take((size_t)CT * EC * 2);
  float* pooled = (float*)take((size_t)BATCH * NPOS * EC * 4);
  float* kbuf   = (float*)take((size_t)BATCH * NPOS * EC * 4);
  float* vbuf   = (float*)take((size_t)BATCH * NPOS * EC * 4);
  unsigned short* qbf   = (unsigned short*)take((size_t)BATCH * NTXT * EC * 2);
  unsigned short* attbf = (unsigned short*)take((size_t)BATCH * NTXT * EC * 2);

  // Weight precision conversion (+ transpose for B matrices).
  cvt_kernel<<<256, 256, 0, stream>>>(cw0, wbf0, EC * 256);
  cvt_kernel<<<512, 256, 0, stream>>>(cw1, wbf1, EC * 512);
  cvt_kernel<<<512, 256, 0, stream>>>(cw2, wbf2, EC * 512);
  cvtT_kernel<<<512, 256, 0, stream>>>(qw, qwT, CT * EC, 8, CT);   // [512][256]->[256][512]
  cvtT_kernel<<<512, 256, 0, stream>>>(pw, prT, EC * CT, 9, EC);   // [256][512]->[512][256]

  // Fused conv + max-pool, one block per (window, batch), double-buffered.
  size_t smem0 = (size_t)(256 + 32) * (256 + 8) * 2;   // ~152 KB
  size_t smem1 = (size_t)(256 + 32) * (512 + 8) * 2;   // ~300 KB (<320KB WGP LDS)
  convpool_kernel<256, 96, 32><<<dim3(9, BATCH), 256, smem0, stream>>>(
      x0, wbf0, cb0, pooled, 0);
  convpool_kernel<512, 48, 16><<<dim3(9, BATCH), 256, smem1, stream>>>(
      x1, wbf1, cb1, pooled, 9);
  convpool_kernel<512, 24, 8><<<dim3(9, BATCH), 256, smem1, stream>>>(
      x2, wbf2, cb2, pooled, 18);

  kv_kernel<<<BATCH, 256, 0, stream>>>(pooled, klg, klb, kw, kb, vlg, vlb, vw, vb, kbuf, vbuf);
  q_kernel<<<dim3(NTXT / 64, BATCH), 256, (size_t)64 * (CT + 8) * 2, stream>>>(
      txt, qlg, qlb, qwT, qb, qbf);
  attn_kernel<<<BATCH, 256, 0, stream>>>(qbf, kbuf, vbuf, attbf);
  proj_kernel<<<dim3(NTXT / 32, BATCH), 256, 0, stream>>>(attbf, prT, pb, txt, out);
}
